// ExpertLoRA_31568009625805
// MI455X (gfx1250) — matrix operand order repaired and checked by reading, not compiled, not run on hardware
//
#include <hip/hip_runtime.h>

// ---------------- problem constants (from reference) ----------------
#define E_ 8
#define H_ 2048
#define F_ 1024
#define D_ 2048          // 2*F
#define T_ 2048          // B*S
#define K_ 4
#define R_ 8
#define SCALING_ 2.0f    // LORA_ALPHA / R
#define LIMIT_ 7.0f
#define ALPHA_ 1.702f

typedef unsigned short u16;
typedef unsigned int   u32;
typedef __attribute__((ext_vector_type(16))) __bf16 v16bf;
typedef __attribute__((ext_vector_type(8)))  float  v8f;

union Frag { v16bf v; uint4 q[2]; };

__device__ __forceinline__ u16 f2bf(float f) {
  union { float f; u32 u; } x; x.f = f;
  u32 r = x.u + 0x7FFFu + ((x.u >> 16) & 1u);   // round-to-nearest-even
  return (u16)(r >> 16);
}

__device__ __forceinline__ v8f v8f_zero() {
  v8f z;
#pragma unroll
  for (int i = 0; i < 8; ++i) z[i] = 0.0f;
  return z;
}

// ---------------- routing: per-expert weight matrix + compact token lists ----
__global__ void zero_counts_kernel(int* counts) {
  if (threadIdx.x < E_) counts[threadIdx.x] = 0;
}

__global__ void routing_kernel(const float* __restrict__ rw, const int* __restrict__ ridx,
                               float* __restrict__ wET, int* __restrict__ counts,
                               int* __restrict__ lists) {
  int t = blockIdx.x * blockDim.x + threadIdx.x;
  if (t >= T_) return;
  float acc[E_];
#pragma unroll
  for (int j = 0; j < E_; ++j) acc[j] = 0.0f;
#pragma unroll
  for (int k = 0; k < K_; ++k) {
    int e = ridx[t * K_ + k];
    float v = rw[t * K_ + k];
#pragma unroll
    for (int j = 0; j < E_; ++j) acc[j] += (j == e) ? v : 0.0f;
  }
#pragma unroll
  for (int j = 0; j < E_; ++j) {
    wET[j * T_ + t] = acc[j];
    if (acc[j] != 0.0f) {                 // token contributes to expert j
      int slot = atomicAdd(&counts[j], 1);
      lists[j * T_ + slot] = t;
    }
  }
}

// ---------------- fp32 x -> bf16 x ----------------
__global__ void convert_x_kernel(const float* __restrict__ x, u16* __restrict__ xb) {
  int i = blockIdx.x * blockDim.x + threadIdx.x;
  if (i * 4 >= T_ * H_) return;
  float4 v = ((const float4*)x)[i];
  ushort4 o;
  o.x = f2bf(v.x); o.y = f2bf(v.y); o.z = f2bf(v.z); o.w = f2bf(v.w);
  ((ushort4*)xb)[i] = o;
}

// ------- weight prep: W' = W + SCALING*A@B, transposed to [e][n][k], bf16 -------
// w: [E][Kd][Nd], A: [E][Kd][R], Bm: [E][R][Nd] -> wt: [E][Nd][Kd]
__global__ __launch_bounds__(256)
void prep_w_kernel(const float* __restrict__ w, const float* __restrict__ A,
                   const float* __restrict__ Bm, u16* __restrict__ wt,
                   int Kd, int Nd) {
  __shared__ float tile[32][33];
  int e  = blockIdx.z;
  int k0 = blockIdx.y * 32, n0 = blockIdx.x * 32;
  int tx = threadIdx.x, ty = threadIdx.y;   // blockDim = (32,8)
#pragma unroll
  for (int j = 0; j < 4; ++j) {
    int k = k0 + ty + j * 8;
    int n = n0 + tx;
    float v = w[((size_t)e * Kd + k) * Nd + n];
    float s = 0.0f;
#pragma unroll
    for (int r = 0; r < R_; ++r)
      s += A[((size_t)e * Kd + k) * R_ + r] * Bm[((size_t)e * R_ + r) * Nd + n];
    tile[ty + j * 8][tx] = v + SCALING_ * s;
  }
  __syncthreads();
#pragma unroll
  for (int j = 0; j < 4; ++j) {
    int n = n0 + ty + j * 8;
    int k = k0 + tx;
    wt[((size_t)e * Nd + n) * Kd + k] = f2bf(tile[tx][ty + j * 8]);
  }
}

// ---------------- out init: routed-weighted down-bias ----------------
__global__ void init_out_kernel(const float* __restrict__ wET,
                                const float* __restrict__ bias2,
                                float* __restrict__ out) {
  int i = blockIdx.x * blockDim.x + threadIdx.x;
  if (i >= T_ * H_) return;
  int t = i >> 11;          // / H_
  int h = i & (H_ - 1);
  float s = 0.0f;
#pragma unroll
  for (int e = 0; e < E_; ++e) s += wET[e * T_ + t] * bias2[e * H_ + h];
  out[i] = s;
}

// ---------------- phase 1: gate_up GEMM (bf16 WMMA) + GLU -> gated ----------
// block = 8 waves -> tile 64 tokens x 256 D-cols; wave tile 32x64 (2x4 WMMA frags)
__global__ __launch_bounds__(256)
void phase1_kernel(const u16* __restrict__ xb, const u16* __restrict__ w1t,
                   const float* __restrict__ gbias, const int* __restrict__ lists,
                   const int* __restrict__ counts, u16* __restrict__ gated) {
  const int e   = blockIdx.z;
  const int cnt = counts[e];
  const int m0  = blockIdx.y * 64;
  if (m0 >= cnt) return;
  const int n0   = blockIdx.x * 256;
  const int lane = threadIdx.x & 31;
  const int wv   = threadIdx.x >> 5;
  const int wm   = wv & 1;
  const int wn   = wv >> 1;
  const int half = lane >> 4;
  const int lr   = lane & 15;
  const int* list = lists + e * T_;

  const u16* arow[2];
#pragma unroll
  for (int mi = 0; mi < 2; ++mi) {
    int m  = m0 + wm * 32 + mi * 16 + lr;
    int mc = m < cnt ? m : cnt - 1;
    arow[mi] = xb + (size_t)list[mc] * H_;
  }
  const u16* bptr[4];
  int ncol[4];
#pragma unroll
  for (int ni = 0; ni < 4; ++ni) {
    int n = n0 + wn * 64 + ni * 16 + lr;
    ncol[ni] = n;
    bptr[ni] = w1t + ((size_t)e * D_ + n) * H_ + half * 16;
  }

  v8f acc[2][4];
#pragma unroll
  for (int mi = 0; mi < 2; ++mi)
#pragma unroll
    for (int ni = 0; ni < 4; ++ni) acc[mi][ni] = v8f_zero();

  for (int k0 = 0; k0 < H_; k0 += 32) {
    Frag a[2], b[4];
#pragma unroll
    for (int mi = 0; mi < 2; ++mi) {
      const uint4* p = (const uint4*)(arow[mi] + k0);
      a[mi].q[0] = p[2 * half];    // K blocks {0..7 | 16..23} per lane-half
      a[mi].q[1] = p[2 * half + 1]; // K blocks {8..15 | 24..31}
    }
#pragma unroll
    for (int ni = 0; ni < 4; ++ni) {
      const uint4* p = (const uint4*)(bptr[ni] + k0);
      b[ni].q[0] = p[0];           // 16 consecutive K (N-major weights)
      b[ni].q[1] = p[1];
    }
#pragma unroll
    for (int mi = 0; mi < 2; ++mi)
#pragma unroll
      for (int ni = 0; ni < 4; ++ni)
        acc[mi][ni] = __builtin_amdgcn_wmma_f32_16x16x32_bf16(
            false, a[mi].v, false, b[ni].v, (short)0, acc[mi][ni], false, false);
  }

  // epilogue: + bias, pair even/odd D columns via lane-xor shuffle, GLU
#pragma unroll
  for (int mi = 0; mi < 2; ++mi) {
#pragma unroll
    for (int ni = 0; ni < 4; ++ni) {
      int n = ncol[ni];
      float bv = gbias[e * D_ + n];
#pragma unroll
      for (int r = 0; r < 8; ++r) {
        int m = m0 + wm * 32 + mi * 16 + half * 8 + r;
        float val = acc[mi][ni][r] + bv;
        float other = __shfl_xor(val, 1, 32);
        if (((n & 1) == 0) && (m < cnt)) {
          float g = fminf(val, LIMIT_);
          float u = fminf(fmaxf(other, -LIMIT_), LIMIT_);
          float glu = g / (1.0f + __expf(-ALPHA_ * g));
          float res = (u + 1.0f) * glu;
          gated[((size_t)e * T_ + m) * F_ + (n >> 1)] = f2bf(res);
        }
      }
    }
  }
}

// ---------------- phase 2: down GEMM (bf16 WMMA), weighted accumulate --------
// launched once per expert (sequential on stream) -> deterministic out +=
__global__ __launch_bounds__(256)
void phase2_kernel(const u16* __restrict__ gated, const u16* __restrict__ w2t,
                   const int* __restrict__ lists, const int* __restrict__ counts,
                   const float* __restrict__ wET, float* __restrict__ out, int e) {
  const int cnt = counts[e];
  const int m0  = blockIdx.y * 64;
  if (m0 >= cnt) return;
  const int n0   = blockIdx.x * 256;
  const int lane = threadIdx.x & 31;
  const int wv   = threadIdx.x >> 5;
  const int wm   = wv & 1;
  const int wn   = wv >> 1;
  const int half = lane >> 4;
  const int lr   = lane & 15;
  const int* list = lists + e * T_;

  const u16* arow[2];
#pragma unroll
  for (int mi = 0; mi < 2; ++mi) {
    int m  = m0 + wm * 32 + mi * 16 + lr;
    int mc = m < cnt ? m : cnt - 1;
    arow[mi] = gated + ((size_t)e * T_ + mc) * F_;
  }
  const u16* bptr[4];
  int ncol[4];
#pragma unroll
  for (int ni = 0; ni < 4; ++ni) {
    int n = n0 + wn * 64 + ni * 16 + lr;
    ncol[ni] = n;
    bptr[ni] = w2t + ((size_t)e * H_ + n) * F_ + half * 16;
  }

  v8f acc[2][4];
#pragma unroll
  for (int mi = 0; mi < 2; ++mi)
#pragma unroll
    for (int ni = 0; ni < 4; ++ni) acc[mi][ni] = v8f_zero();

  for (int k0 = 0; k0 < F_; k0 += 32) {
    Frag a[2], b[4];
#pragma unroll
    for (int mi = 0; mi < 2; ++mi) {
      const uint4* p = (const uint4*)(arow[mi] + k0);
      a[mi].q[0] = p[2 * half];
      a[mi].q[1] = p[2 * half + 1];
    }
#pragma unroll
    for (int ni = 0; ni < 4; ++ni) {
      const uint4* p = (const uint4*)(bptr[ni] + k0);
      b[ni].q[0] = p[0];
      b[ni].q[1] = p[1];
    }
#pragma unroll
    for (int mi = 0; mi < 2; ++mi)
#pragma unroll
      for (int ni = 0; ni < 4; ++ni)
        acc[mi][ni] = __builtin_amdgcn_wmma_f32_16x16x32_bf16(
            false, a[mi].v, false, b[ni].v, (short)0, acc[mi][ni], false, false);
  }

#pragma unroll
  for (int mi = 0; mi < 2; ++mi) {
#pragma unroll
    for (int r = 0; r < 8; ++r) {
      int m = m0 + wm * 32 + mi * 16 + half * 8 + r;
      if (m < cnt) {
        int t = list[m];
        float wt = wET[e * T_ + t];
#pragma unroll
        for (int ni = 0; ni < 4; ++ni) {
          size_t oi = (size_t)t * H_ + ncol[ni];
          out[oi] += acc[mi][ni][r] * wt;   // unique (t,n) per launch
        }
      }
    }
  }
}

// ---------------- host-side orchestration ----------------
extern "C" void kernel_launch(void* const* d_in, const int* in_sizes, int n_in,
                              void* d_out, int out_size, void* d_ws, size_t ws_size,
                              hipStream_t stream) {
  const float* x    = (const float*)d_in[0];   // [B,S,H]
  const float* rw   = (const float*)d_in[1];   // [T,K]
  const float* gup  = (const float*)d_in[2];   // [E,H,D]
  const float* gupb = (const float*)d_in[3];   // [E,D]
  const float* dwp  = (const float*)d_in[4];   // [E,F,H]
  const float* dwb  = (const float*)d_in[5];   // [E,H]
  const float* a1   = (const float*)d_in[6];   // [E,H,R]
  const float* b1   = (const float*)d_in[7];   // [E,R,D]
  const float* a2   = (const float*)d_in[8];   // [E,F,R]
  const float* b2   = (const float*)d_in[9];   // [E,R,H]
  const int*   ridx = (const int*)d_in[10];    // [T,K]
  float* out = (float*)d_out;

  char* ws = (char*)d_ws;
  int*   counts = (int*)ws;                                  // 256 B
  int*   lists  = (int*)(ws + 256);                          // E*T*4   = 64 KiB
  float* wET    = (float*)(ws + 256 + 64 * 1024);            // E*T*4   = 64 KiB
  u16*   xb     = (u16*)(ws + 256 + 128 * 1024);             // T*H*2   = 8 MiB
  u16*   w1t    = xb  + (size_t)T_ * H_;                     // E*D*H*2 = 64 MiB
  u16*   w2t    = w1t + (size_t)E_ * D_ * H_;                // E*H*F*2 = 32 MiB
  u16*   gatedb = w2t + (size_t)E_ * H_ * F_;                // E*T*F*2 = 32 MiB

  zero_counts_kernel<<<1, 32, 0, stream>>>(counts);
  routing_kernel<<<T_ / 256, 256, 0, stream>>>(rw, ridx, wET, counts, lists);
  convert_x_kernel<<<(T_ * H_ / 4) / 256, 256, 0, stream>>>(x, xb);
  prep_w_kernel<<<dim3(D_ / 32, H_ / 32, E_), dim3(32, 8), 0, stream>>>(gup, a1, b1, w1t, H_, D_);
  prep_w_kernel<<<dim3(H_ / 32, F_ / 32, E_), dim3(32, 8), 0, stream>>>(dwp, a2, b2, w2t, F_, H_);
  init_out_kernel<<<(T_ * H_) / 256, 256, 0, stream>>>(wET, dwb, out);
  phase1_kernel<<<dim3(D_ / 256, T_ / 64, E_), 256, 0, stream>>>(xb, w1t, gupb, lists, counts, gatedb);
  for (int e = 0; e < E_; ++e)
    phase2_kernel<<<dim3(H_ / 256, T_ / 64, 1), 256, 0, stream>>>(gatedb, w2t, lists, counts, wET, out, e);
}
